// EnhancedFastKANLayer_58428735095134
// MI455X (gfx1250) — compile-verified, matrix operand order checked by reading, not run on hardware
//
#include <hip/hip_runtime.h>
#include <stdint.h>

typedef __attribute__((ext_vector_type(16))) _Float16 v16h;
typedef __attribute__((ext_vector_type(8)))  _Float16 v8h;
typedef __attribute__((ext_vector_type(8)))  float    v8f;
typedef __attribute__((ext_vector_type(4)))  float    v4f;
typedef __attribute__((ext_vector_type(4)))  int      v4i;

#define BATCH 16384
#define DIM   512
#define KSPL  4096
#define KTOT  4608   // 4096 spline + 512 base
#define NBLK  (KTOT / 64)
#define BM    128
#define BN    128
#define BK    64
#define AST   72     // LDS row stride in f16 (64 + 8 pad -> conflict-free b128)

#define GLOBAL_AS __attribute__((address_space(1)))
#define LDS_AS    __attribute__((address_space(3)))

#if defined(__has_builtin)
#  if __has_builtin(__builtin_amdgcn_global_load_async_to_lds_b128)
#    define HAVE_ASYNC_LDS 1
#  endif
#endif

#if defined(HAVE_ASYNC_LDS)
#  if __has_builtin(__builtin_amdgcn_s_wait_asynccnt)
#    define ASYNC_WAIT(n) __builtin_amdgcn_s_wait_asynccnt(n)
#  else
#    define ASYNC_WAIT(n) asm volatile("s_wait_asynccnt %0" ::"i"(n) : "memory")
#  endif
#endif

// ---------------------------------------------------------------------------
// Kernel 1: pack [W_spline (4096x512); W_base (512x512)] fp32 into
// Wt[512][4608] f16 (N-major) so B tiles are contiguous 128B rows.
// ---------------------------------------------------------------------------
__global__ __launch_bounds__(256)
void prep_w_kernel(const float* __restrict__ Ws, const float* __restrict__ Wb,
                   _Float16* __restrict__ Wt) {
  __shared__ __align__(16) _Float16 tile[64][AST];
  const int t  = threadIdx.x;
  const int k0 = blockIdx.x * 64;   // 72 tiles
  const int n0 = blockIdx.y * 64;   // 8 tiles

  {
    const int nn  = (t & 15) * 4;
    const int kk0 = t >> 4;
#pragma unroll
    for (int r = 0; r < 4; ++r) {
      const int kk = kk0 + r * 16;
      const int kg = k0 + kk;
      const float* src = (kg < KSPL)
          ? (Ws + (size_t)kg * DIM + n0 + nn)
          : (Wb + (size_t)(kg - KSPL) * DIM + n0 + nn);
      v4f v = *(const v4f*)src;
      tile[kk][nn + 0] = (_Float16)v.x;
      tile[kk][nn + 1] = (_Float16)v.y;
      tile[kk][nn + 2] = (_Float16)v.z;
      tile[kk][nn + 3] = (_Float16)v.w;
    }
  }
  __syncthreads();
  {
    const int n  = t >> 2;          // 0..63
    const int kq = (t & 3) * 16;    // 0,16,32,48
    v8h a, b;
#pragma unroll
    for (int i = 0; i < 8; ++i) {
      a[i] = tile[kq + i][n];
      b[i] = tile[kq + 8 + i][n];
    }
    _Float16* dst = Wt + (size_t)(n0 + n) * KTOT + k0 + kq;
    *(v8h*)dst       = a;
    *(v8h*)(dst + 8) = b;
  }
}

// ---------------------------------------------------------------------------
__device__ __forceinline__ v16h frag16(const _Float16* rowbase, int kc) {
  v8h lo = *(const v8h*)(rowbase + kc);
  v8h hi = *(const v8h*)(rowbase + kc + 16);
  return __builtin_shufflevector(lo, hi, 0,1,2,3,4,5,6,7,8,9,10,11,12,13,14,15);
}

__device__ __forceinline__ void fill_a_tile(_Float16 (&As)[BM][AST],
                                            const float* __restrict__ x,
                                            const float* bns, const float* bnb,
                                            int kb, int m0, int t) {
  const float gstep = 4.0f / 7.0f;  // linspace(-2,2,8) step
  if (kb < KSPL / BK) {
    // RBF basis block: x cols [kb*8, kb*8+8), 8 grids each
    const int m  = t >> 1;
    const int cg = (t & 1) * 4;
    const int c0 = kb * 8 + cg;
    v4f xv = *(const v4f*)(x + (size_t)(m0 + m) * DIM + c0);
#pragma unroll
    for (int i = 0; i < 4; ++i) {
      const int c = c0 + i;
      const float xn = xv[i] * bns[c] + bnb[c];
      v8h e;
#pragma unroll
      for (int g = 0; g < 8; ++g) {
        const float d = (xn - (-2.0f + g * gstep)) * 2.0f;  // /denom(0.5)
        e[g] = (_Float16)__expf(-d * d);
      }
      *(v8h*)&As[m][(cg + i) * 8] = e;
    }
  } else {
    // base block: relu(x), x cols [(kb-64)*64, +64)
    const int m  = t >> 1;
    const int hh = (t & 1) * 32;
    const float* xr = x + (size_t)(m0 + m) * DIM + (kb - KSPL / BK) * 64 + hh;
#pragma unroll
    for (int j = 0; j < 32; j += 8) {
      v4f a = *(const v4f*)(xr + j);
      v4f b = *(const v4f*)(xr + j + 4);
      v8h e;
#pragma unroll
      for (int q = 0; q < 4; ++q) {
        e[q]     = (_Float16)fmaxf(a[q], 0.0f);
        e[q + 4] = (_Float16)fmaxf(b[q], 0.0f);
      }
      *(v8h*)&As[m][hh + j] = e;
    }
  }
}

#if defined(HAVE_ASYNC_LDS)
// Issue 4 async b128 global->LDS copies (64B per thread) for one B tile.
__device__ __forceinline__ void fill_b_async(_Float16 (&Bs)[BN][AST],
                                             const _Float16* __restrict__ Wt,
                                             int kb, int n0, int t) {
  const int n  = t >> 1;
  const int kh = (t & 1) * 32;
  const _Float16* src = Wt + (size_t)(n0 + n) * KTOT + (size_t)kb * BK + kh;
  _Float16* dst = &Bs[n][kh];
#pragma unroll
  for (int j = 0; j < 32; j += 8)
    __builtin_amdgcn_global_load_async_to_lds_b128(
        (GLOBAL_AS v4i*)(GLOBAL_AS void*)(src + j),
        (LDS_AS v4i*)(LDS_AS void*)(dst + j), 0, 0);
}
#else
__device__ __forceinline__ void fill_b_manual(_Float16 (&Bs)[BN][AST],
                                              const _Float16* __restrict__ Wt,
                                              int kb, int n0, int t) {
  const int n  = t >> 1;
  const int kh = (t & 1) * 32;
  const _Float16* src = Wt + (size_t)(n0 + n) * KTOT + (size_t)kb * BK + kh;
  _Float16* dst = &Bs[n][kh];
#pragma unroll
  for (int j = 0; j < 32; j += 8)
    *(v8h*)(dst + j) = *(const v8h*)(src + j);
}
#endif

__device__ __forceinline__ void compute_tile(v8f (&acc)[4][2],
                                             const _Float16 (&As)[BM][AST],
                                             const _Float16 (&Bs)[BN][AST],
                                             int mbase, int nbase, int lrow, int khi) {
#pragma unroll
  for (int ks = 0; ks < 2; ++ks) {
    const int kc = ks * 32 + khi;
    v16h bf[2], af[4];
#pragma unroll
    for (int j = 0; j < 2; ++j)
      bf[j] = frag16(&Bs[nbase + j * 16 + lrow][0], kc);
#pragma unroll
    for (int i = 0; i < 4; ++i)
      af[i] = frag16(&As[mbase + i * 16 + lrow][0], kc);
#pragma unroll
    for (int i = 0; i < 4; ++i)
#pragma unroll
      for (int j = 0; j < 2; ++j)
        acc[i][j] = __builtin_amdgcn_wmma_f32_16x16x32_f16(
            false, af[i], false, bf[j], (short)0, acc[i][j], false, false);
  }
}

// ---------------------------------------------------------------------------
// Kernel 2: fused BN -> RBF basis -> (basis @ Wspl + relu(x) @ Wbase) WMMA GEMM
// with async-prefetched (double-buffered) B tiles, bias + residual epilogue.
// Grid: (16384/128, 512/128) = (128, 4), 256 threads = 8 waves.
// ---------------------------------------------------------------------------
__global__ __launch_bounds__(256)
void fastkan_main_kernel(const float* __restrict__ x,
                         const float* __restrict__ bn_gamma,
                         const float* __restrict__ bn_beta,
                         const float* __restrict__ bn_mean,
                         const float* __restrict__ bn_var,
                         const _Float16* __restrict__ Wt,
                         const float* __restrict__ b_spl,
                         const float* __restrict__ b_base,
                         float* __restrict__ out) {
  __shared__ __align__(16) _Float16 As[BM][AST];
  __shared__ __align__(16) _Float16 Bs[2][BN][AST];
  __shared__ float bns[DIM];
  __shared__ float bnb[DIM];

  const int t  = threadIdx.x;
  const int m0 = blockIdx.x * BM;
  const int n0 = blockIdx.y * BN;

  for (int c = t; c < DIM; c += 256) {
    float s = bn_gamma[c] * rsqrtf(bn_var[c] + 1e-3f);
    bns[c] = s;
    bnb[c] = bn_beta[c] - bn_mean[c] * s;
  }

  const int lane  = t & 31;
  const int wave  = t >> 5;
  const int mbase = (wave & 1) * 64;
  const int nbase = (wave >> 1) * 32;
  const int lrow  = lane & 15;
  const int khi   = (lane >> 4) * 8;

  v8f acc[4][2] = {};

#if defined(HAVE_ASYNC_LDS)
  fill_b_async(Bs[0], Wt, 0, n0, t);  // prefetch first B tile
#endif

  for (int kb = 0; kb < NBLK; ++kb) {
    const int cur = kb & 1;
    __syncthreads();  // prev compute done: As and Bs[cur^1] are free

    fill_a_tile(As, x, bns, bnb, kb, m0, t);

#if defined(HAVE_ASYNC_LDS)
    if (kb + 1 < NBLK) {
      fill_b_async(Bs[cur ^ 1], Wt, kb + 1, n0, t);
      ASYNC_WAIT(4);  // in-order completion: tile kb's 4 copies are done
    } else {
      ASYNC_WAIT(0);
    }
#else
    fill_b_manual(Bs[cur], Wt, kb, n0, t);
#endif

    __syncthreads();
    compute_tile(acc, As, Bs[cur], mbase, nbase, lrow, khi);
  }

  // ---- epilogue: + b_spline + b_base + residual x ----
#pragma unroll
  for (int j = 0; j < 2; ++j) {
    const int n = n0 + nbase + j * 16 + lrow;       // C: N = lane&15
    const float bias = b_spl[n] + b_base[n];
#pragma unroll
    for (int i = 0; i < 4; ++i) {
      const int mb = m0 + mbase + i * 16 + khi;     // C: M = r + 8*(lane>>4)
#pragma unroll
      for (int r = 0; r < 8; ++r) {
        const size_t off = (size_t)(mb + r) * DIM + n;
        out[off] = acc[i][j][r] + bias + x[off];
      }
    }
  }
}

// ---------------------------------------------------------------------------
extern "C" void kernel_launch(void* const* d_in, const int* in_sizes, int n_in,
                              void* d_out, int out_size, void* d_ws, size_t ws_size,
                              hipStream_t stream) {
  (void)in_sizes; (void)n_in; (void)out_size; (void)ws_size;
  const float* x      = (const float*)d_in[0];
  const float* gammap = (const float*)d_in[1];
  const float* betap  = (const float*)d_in[2];
  const float* meanp  = (const float*)d_in[3];
  const float* varp   = (const float*)d_in[4];
  const float* Wspl   = (const float*)d_in[5];
  const float* bspl   = (const float*)d_in[6];
  const float* Wbase  = (const float*)d_in[7];
  const float* bbase  = (const float*)d_in[8];
  float* out = (float*)d_out;

  _Float16* Wt = (_Float16*)d_ws;   // 512*4608*2 = 4.5 MB scratch

  prep_w_kernel<<<dim3(KTOT / 64, DIM / 64), 256, 0, stream>>>(Wspl, Wbase, Wt);
  fastkan_main_kernel<<<dim3(BATCH / BM, DIM / BN), 256, 0, stream>>>(
      x, gammap, betap, meanp, varp, Wt, bspl, bbase, out);
}
